// DevignModel_63247688401221
// MI455X (gfx1250) — compile-verified, matrix operand order checked by reading, not run on hardware
//
#include <hip/hip_runtime.h>
#include <hip/hip_bf16.h>

// ---------------- problem constants ----------------
#define NB     32
#define NNODE  513
#define NTOT   (NB * NNODE)     // 16416
#define INF_   128
#define OUTF   256
#define CATF   384
#define NTYPE  4
#define NSTEP  4
#define NEDGE  (NTOT * 16)      // 262656

typedef __bf16 bf16x16 __attribute__((ext_vector_type(16)));
typedef float  f32x8   __attribute__((ext_vector_type(8)));

__device__ __forceinline__ __bf16 f2bf(float f) {
  unsigned u = __builtin_bit_cast(unsigned, f);
  u += 0x7FFFu + ((u >> 16) & 1u);            // round-to-nearest-even
  unsigned short s = (unsigned short)(u >> 16);
  return __builtin_bit_cast(__bf16, s);
}

union Frag {                                   // 32B fragment = two b128 loads
  bf16x16 f;
  uint4   q[2];
};

// ----------------------------------------------------------------------
// Register-blocked WMMA GEMM on pre-converted bf16 operands.
// One wave computes a 32x64 macro-tile of C (2 M-tiles x 4 N-tiles):
// per 32-wide K slab, 2 A frags + 4 B frags feed 8 v_wmma_f32_16x16x32_bf16
// (0.75KB of loads per WMMA vs 2KB for an unblocked 16x16 tile).
//   C[vr, n] = act( sum_tap sum_k A[amap(vr)+tap, k] * Wp[(n*taps+tap)*K + k]
//                   + bias[n] )
// A-row remap: vr -> g = vr/outRPG, l = vr%outRPG, row = g*inRPG + l
// (implicit-GEMM Conv1d: tap shifts the A row; Wp is pre-packed [Nc,taps,K])
// Fragment layout per CDNA5 ISA 7.12.2 (16-bit, wave32): lane L holds
// row/col (L&15); K pairs {hi*8+0..7} in VGPRs 0-3, {16+hi*8+0..7} in 4-7.
// Both halves are 16B-contiguous in memory -> uint4 loads, zero cvt VALU.
// ----------------------------------------------------------------------
template <int KV>
__global__ void __launch_bounds__(256)
wmma_gemm_bf16(const __bf16* __restrict__ A, const __bf16* __restrict__ Wp,
               const float* __restrict__ bias, float* __restrict__ C,
               int Mrows, int Nc, int lda, int ntaps,
               int outRPG, int inRPG, int relu)
{
  const int wave        = blockIdx.x * (blockDim.x >> 5) + (threadIdx.x >> 5);
  const int tiles_n     = Nc >> 6;                     // 64-wide N macro-tiles
  const int tiles_total = (Mrows >> 5) * tiles_n;      // 32-tall M macro-tiles
  if (wave >= tiles_total) return;                     // wave-uniform: EXEC all-ones
  const int tm   = wave / tiles_n;
  const int tn   = wave - tm * tiles_n;
  const int lane = threadIdx.x & 31;
  const int lr   = lane & 15;
  const int hi   = lane >> 4;

  // A-side source rows for this lane (group remap for conv batching)
  long arow0[2];
#pragma unroll
  for (int mi = 0; mi < 2; ++mi) {
    const int vr = tm * 32 + mi * 16 + lr;
    const int g  = vr / outRPG;
    const int l  = vr - g * outRPG;
    arow0[mi] = (long)g * inRPG + l;
  }
  // B-side base column for this lane
  const int ncol0 = tn * 64 + lr;

  f32x8 acc[2][4];
#pragma unroll
  for (int mi = 0; mi < 2; ++mi)
#pragma unroll
    for (int j = 0; j < 4; ++j)
      acc[mi][j] = (f32x8){0.f, 0.f, 0.f, 0.f, 0.f, 0.f, 0.f, 0.f};

  for (int tap = 0; tap < ntaps; ++tap) {
    const __bf16* __restrict__ arow[2] = {
        A + (arow0[0] + tap) * (long)lda + hi * 8,
        A + (arow0[1] + tap) * (long)lda + hi * 8};
#pragma unroll
    for (int kk = 0; kk < KV; kk += 32) {
      if (kk + 32 < KV) {                              // next K slab -> global_prefetch_b8
        __builtin_prefetch(arow[0] + kk + 32, 0, 1);
        __builtin_prefetch(arow[1] + kk + 32, 0, 1);
      }
      Frag fa[2];
#pragma unroll
      for (int mi = 0; mi < 2; ++mi) {
        fa[mi].q[0] = *(const uint4*)(arow[mi] + kk);
        fa[mi].q[1] = *(const uint4*)(arow[mi] + kk + 16);
      }
#pragma unroll
      for (int j = 0; j < 4; ++j) {
        const __bf16* __restrict__ wrow =
            Wp + ((long)(ncol0 + j * 16) * ntaps + tap) * KV + hi * 8 + kk;
        Frag fb;
        fb.q[0] = *(const uint4*)(wrow);
        fb.q[1] = *(const uint4*)(wrow + 16);
        acc[0][j] = __builtin_amdgcn_wmma_f32_16x16x32_bf16(false, fa[0].f, false, fb.f,
                                                            (short)0, acc[0][j], false, false);
        acc[1][j] = __builtin_amdgcn_wmma_f32_16x16x32_bf16(false, fa[1].f, false, fb.f,
                                                            (short)0, acc[1][j], false, false);
      }
    }
  }

  // C layout (ISA 7.12.2): VGPR r -> row mi*16 + hi*8 + r, col = ncol0 + j*16
#pragma unroll
  for (int mi = 0; mi < 2; ++mi) {
#pragma unroll
    for (int j = 0; j < 4; ++j) {
      const int   col = ncol0 + j * 16;
      const float bv  = bias ? bias[col] : 0.0f;
#pragma unroll
      for (int r = 0; r < 8; ++r) {
        const int crow = tm * 32 + mi * 16 + hi * 8 + r;
        float v = acc[mi][j][r] + bv;
        if (relu) v = fmaxf(v, 0.0f);
        C[(long)crow * Nc + col] = v;
      }
    }
  }
}

// ---------------- conversion / packing kernels ----------------
// straight f32 -> bf16 (n multiple of 4)
__global__ void cvt_bf16(const float* __restrict__ in, __bf16* __restrict__ outp, long n) {
  long i = ((long)blockIdx.x * blockDim.x + threadIdx.x) * 4;
  if (i >= n) return;
  float4 v = *(const float4*)(in + i);
  union { __bf16 b[4]; uint2 u; } o;
  o.b[0] = f2bf(v.x); o.b[1] = f2bf(v.y); o.b[2] = f2bf(v.z); o.b[3] = f2bf(v.w);
  *(uint2*)(outp + i) = o.u;
}

// repack conv weights [Co, Ci, Kw] (f32) -> [Co, Kw, Ci] (bf16) for unit-stride K
__global__ void pack_w(const float* __restrict__ W, __bf16* __restrict__ P,
                       int Nc, int K, int wsN, int wsK, int taps) {
  long i   = (long)blockIdx.x * blockDim.x + threadIdx.x;
  long tot = (long)Nc * taps * K;
  if (i >= tot) return;
  int  k   = (int)(i % K);
  long t2  = i / K;
  int  tap = (int)(t2 % taps);
  int  n   = (int)(t2 / taps);
  P[i] = f2bf(W[(long)n * wsN + (long)k * wsK + tap]);
}

// ---------------- elementwise / data movement kernels ----------------
__global__ void zero_f4(float4* __restrict__ p, long n4) {
  long i = (long)blockIdx.x * blockDim.x + threadIdx.x;
  if (i < n4) p[i] = make_float4(0.f, 0.f, 0.f, 0.f);
}

__global__ void pad_feat(const float* __restrict__ f, float* __restrict__ h) {
  long i = (long)blockIdx.x * blockDim.x + threadIdx.x;
  if (i >= (long)NTOT * OUTF) return;
  int  c = (int)(i & (OUTF - 1));
  long n = i >> 8;
  h[i] = (c < INF_) ? f[n * INF_ + c] : 0.0f;
}

// cat = [h | feature] written directly as bf16 (GEMM input only)
__global__ void concat_bf16(const float* __restrict__ h, const float* __restrict__ f,
                            __bf16* __restrict__ catb) {
  long i = (long)blockIdx.x * blockDim.x + threadIdx.x;
  if (i >= (long)NTOT * CATF) return;
  long n = i / CATF;
  int  c = (int)(i - n * CATF);
  catb[i] = f2bf((c < OUTF) ? h[n * OUTF + c] : f[n * INF_ + (c - OUTF)]);
}

// scatter-add messages of edge-type t:  a[dst] += hwt[src]  (hw f32 atomics)
__global__ void edge_scatter(const float* __restrict__ hwt,
                             const int* __restrict__ esrc, const int* __restrict__ edst,
                             const int* __restrict__ etyp, int t,
                             float* __restrict__ a)
{
  int idx = blockIdx.x * blockDim.x + threadIdx.x;
  int e   = idx >> 2;
  if (e >= NEDGE) return;
  if (etyp[e] != t) return;
  int c0 = (idx & 3) * 64;
  const float4* __restrict__ src = (const float4*)(hwt + (long)esrc[e] * OUTF + c0);
  float* __restrict__ dst = a + (long)edst[e] * OUTF + c0;
#pragma unroll
  for (int i = 0; i < 16; ++i) {
    float4 v = src[i];
    __hip_atomic_fetch_add(dst + 4 * i + 0, v.x, __ATOMIC_RELAXED, __HIP_MEMORY_SCOPE_AGENT);
    __hip_atomic_fetch_add(dst + 4 * i + 1, v.y, __ATOMIC_RELAXED, __HIP_MEMORY_SCOPE_AGENT);
    __hip_atomic_fetch_add(dst + 4 * i + 2, v.z, __ATOMIC_RELAXED, __HIP_MEMORY_SCOPE_AGENT);
    __hip_atomic_fetch_add(dst + 4 * i + 3, v.w, __ATOMIC_RELAXED, __HIP_MEMORY_SCOPE_AGENT);
  }
}

// GRUCell (torch gate order r,z,n):  h = (1-z)*n + z*h
__global__ void gru_update(const float* __restrict__ gi, const float* __restrict__ gh,
                           float* __restrict__ h)
{
  long idx = (long)blockIdx.x * blockDim.x + threadIdx.x;
  if (idx >= (long)NTOT * OUTF) return;
  long n  = idx >> 8;
  int  c  = (int)(idx & (OUTF - 1));
  long bi = n * (3 * OUTF) + c;
  float ir = gi[bi], iz = gi[bi + OUTF], in_ = gi[bi + 2 * OUTF];
  float hr = gh[bi], hz = gh[bi + OUTF], hn  = gh[bi + 2 * OUTF];
  float r  = 1.f / (1.f + __expf(-(ir + hr)));
  float z  = 1.f / (1.f + __expf(-(iz + hz)));
  float ng = tanhf(in_ + r * hn);
  h[idx] = (1.f - z) * ng + z * h[idx];
}

// channel-last maxpool over L:  in [G, Lin, C] -> out [G, Lout, C]
__global__ void maxpool_cl(const float* __restrict__ in, float* __restrict__ outp,
                           int G, int Lin, int Lout, int C, int win, int stride)
{
  long idx = (long)blockIdx.x * blockDim.x + threadIdx.x;
  int  C4  = C >> 2;
  long tot = (long)G * Lout * C4;
  if (idx >= tot) return;
  int  c4 = (int)(idx % C4);
  long t  = idx / C4;
  int  lo = (int)(t % Lout);
  int  g  = (int)(t / Lout);
  const float* base = in + ((long)g * Lin + (long)lo * stride) * C + c4 * 4;
  float4 m = *(const float4*)base;
  for (int w = 1; w < win; ++w) {
    float4 v = *(const float4*)(base + (long)w * C);
    m.x = fmaxf(m.x, v.x); m.y = fmaxf(m.y, v.y);
    m.z = fmaxf(m.z, v.z); m.w = fmaxf(m.w, v.w);
  }
  *(float4*)(outp + ((long)g * Lout + lo) * C + c4 * 4) = m;
}

// per-graph readout: sigmoid( mean_l ( (Y2.mlpy + by) * (Z2.mlpz + bz) ) )
__global__ void head_readout(const float* __restrict__ y2p, const float* __restrict__ z2p,
                             const float* __restrict__ myw, const float* __restrict__ myb,
                             const float* __restrict__ mzw, const float* __restrict__ mzb,
                             float* __restrict__ out)
{
  __shared__ float red[128];
  int b = blockIdx.x, tid = threadIdx.x;
  float local = 0.f;
  for (int l = tid; l < 127; l += 128) {
    const float* yr = y2p + ((long)b * 127 + l) * OUTF;
    const float* zr = z2p + ((long)b * 127 + l) * CATF;
    float y = myb[0];
    for (int c = 0; c < OUTF; ++c) y += yr[c] * myw[c];
    float z = mzb[0];
    for (int c = 0; c < CATF; ++c) z += zr[c] * mzw[c];
    local += y * z;
  }
  red[tid] = local;
  __syncthreads();
  for (int s = 64; s > 0; s >>= 1) {
    if (tid < s) red[tid] += red[tid + s];
    __syncthreads();
  }
  if (tid == 0) {
    float avg = red[0] / 127.0f;
    out[b] = 1.0f / (1.0f + __expf(-avg));
  }
}

// ---------------- host-side launch helpers ----------------
static inline int cdiv(long a, int b) { return (int)((a + b - 1) / b); }

static inline void gemm(hipStream_t s, const __bf16* A, const __bf16* Wp, const float* b,
                        float* C, int M, int Nc, int K, int lda, int taps,
                        int oRPG, int iRPG, int relu)
{
  int tiles  = (M / 32) * (Nc / 64);           // 32x64 macro-tiles per wave
  int blocks = (tiles + 7) / 8;                // 8 waves / 256-thread block
  if (K == 256)
    wmma_gemm_bf16<256><<<blocks, 256, 0, s>>>(A, Wp, b, C, M, Nc, lda, taps, oRPG, iRPG, relu);
  else
    wmma_gemm_bf16<384><<<blocks, 256, 0, s>>>(A, Wp, b, C, M, Nc, lda, taps, oRPG, iRPG, relu);
}

extern "C" void kernel_launch(void* const* d_in, const int* in_sizes, int n_in,
                              void* d_out, int out_size, void* d_ws, size_t ws_size,
                              hipStream_t stream)
{
  (void)in_sizes; (void)n_in; (void)out_size; (void)ws_size;
  const float* feature = (const float*)d_in[0];
  const float* W_msg   = (const float*)d_in[1];
  const float* b_msg   = (const float*)d_in[2];
  const float* gw_ih   = (const float*)d_in[3];
  const float* gw_hh   = (const float*)d_in[4];
  const float* gb_ih   = (const float*)d_in[5];
  const float* gb_hh   = (const float*)d_in[6];
  const float* c1w     = (const float*)d_in[7];
  const float* c1b     = (const float*)d_in[8];
  const float* c2w     = (const float*)d_in[9];
  const float* c2b     = (const float*)d_in[10];
  const float* cc1w    = (const float*)d_in[11];
  const float* cc1b    = (const float*)d_in[12];
  const float* cc2w    = (const float*)d_in[13];
  const float* cc2b    = (const float*)d_in[14];
  const float* myw     = (const float*)d_in[15];
  const float* myb     = (const float*)d_in[16];
  const float* mzw     = (const float*)d_in[17];
  const float* mzb     = (const float*)d_in[18];
  const int*   esrc    = (const int*)d_in[19];
  const int*   edst    = (const int*)d_in[20];
  const int*   etyp    = (const int*)d_in[21];
  float*       out     = (float*)d_out;

  // ---- workspace carve-out (256B aligned) ----
  char*  base = (char*)d_ws;
  size_t off  = 0;
  auto alloc = [&](size_t bytes) -> char* {
    char* p = base + off;
    off += (bytes + 255) & ~(size_t)255;
    return p;
  };
  const long NF = NTOT;
  float*  h     = (float*)alloc(NF * OUTF * 4);           // persistent f32 state
  __bf16* hb    = (__bf16*)alloc(NF * OUTF * 2);          // bf16 mirror of h
  __bf16* wmsgb = (__bf16*)alloc((size_t)NTYPE * OUTF * OUTF * 2);
  __bf16* wihb  = (__bf16*)alloc((size_t)768 * OUTF * 2);
  __bf16* whhb  = (__bf16*)alloc((size_t)768 * OUTF * 2);
  __bf16* c1p   = (__bf16*)alloc((size_t)OUTF * 3 * OUTF * 2);
  __bf16* c2p   = (__bf16*)alloc((size_t)OUTF * OUTF * 2);
  __bf16* cc1p  = (__bf16*)alloc((size_t)CATF * 3 * CATF * 2);
  __bf16* cc2p  = (__bf16*)alloc((size_t)CATF * 2 * CATF * 2);
  const size_t paOff = off;
  // phase A (GGNN)
  float*  hwt  = (float*)alloc(NF * OUTF * 4);
  float*  abuf = (float*)alloc(NF * OUTF * 4);
  __bf16* ab   = (__bf16*)alloc(NF * OUTF * 2);
  float*  gi   = (float*)alloc(NF * 768 * 4);
  float*  gh   = (float*)alloc(NF * 768 * 4);
  // phase B (heads) aliases phase-A region
  off = paOff;
  __bf16* catb = (__bf16*)alloc(NF * CATF * 2);
  float*  y1   = (float*)alloc(32L * 511 * OUTF * 4);
  float*  z1   = (float*)alloc(32L * 511 * CATF * 4);
  float*  y1p  = (float*)alloc(32L * 255 * OUTF * 4);
  __bf16* y1pb = (__bf16*)alloc(32L * 255 * OUTF * 2);
  float*  z1p  = (float*)alloc(32L * 255 * CATF * 4);
  __bf16* z1pb = (__bf16*)alloc(32L * 255 * CATF * 2);
  float*  y2   = (float*)alloc(32L * 255 * OUTF * 4);
  float*  z2   = (float*)alloc(32L * 254 * CATF * 4);
  float*  y2p  = (float*)alloc(32L * 127 * OUTF * 4);
  float*  z2p  = (float*)alloc(32L * 127 * CATF * 4);

  // ---- pack weights to bf16 once (cheap, bandwidth-bound) ----
  cvt_bf16<<<cdiv((long)NTYPE * OUTF * OUTF / 4, 256), 256, 0, stream>>>(W_msg, wmsgb, (long)NTYPE * OUTF * OUTF);
  cvt_bf16<<<cdiv(768L * OUTF / 4, 256), 256, 0, stream>>>(gw_ih, wihb, 768L * OUTF);
  cvt_bf16<<<cdiv(768L * OUTF / 4, 256), 256, 0, stream>>>(gw_hh, whhb, 768L * OUTF);
  cvt_bf16<<<cdiv((long)OUTF * OUTF / 4, 256), 256, 0, stream>>>(c2w, c2p, (long)OUTF * OUTF);
  pack_w<<<cdiv((long)OUTF * 3 * OUTF, 256), 256, 0, stream>>>(c1w, c1p, OUTF, OUTF, OUTF * 3, 3, 3);
  pack_w<<<cdiv((long)CATF * 3 * CATF, 256), 256, 0, stream>>>(cc1w, cc1p, CATF, CATF, CATF * 3, 3, 3);
  pack_w<<<cdiv((long)CATF * 2 * CATF, 256), 256, 0, stream>>>(cc2w, cc2p, CATF, CATF, CATF * 2, 2, 2);

  // h = pad(feature, OUT)
  pad_feat<<<cdiv(NF * OUTF, 256), 256, 0, stream>>>(feature, h);

  // ---- GGNN propagation ----
  for (int step = 0; step < NSTEP; ++step) {
    cvt_bf16<<<cdiv(NF * OUTF / 4, 256), 256, 0, stream>>>(h, hb, NF * OUTF);
    zero_f4<<<cdiv(NF * OUTF / 4, 256), 256, 0, stream>>>((float4*)abuf, NF * OUTF / 4);
    for (int t = 0; t < NTYPE; ++t) {
      gemm(stream, hb, wmsgb + (long)t * OUTF * OUTF, b_msg + (long)t * OUTF, hwt,
           NTOT, OUTF, OUTF, OUTF, 1, NTOT, NTOT, 0);
      edge_scatter<<<cdiv((long)NEDGE * 4, 256), 256, 0, stream>>>(hwt, esrc, edst, etyp, t, abuf);
    }
    cvt_bf16<<<cdiv(NF * OUTF / 4, 256), 256, 0, stream>>>(abuf, ab, NF * OUTF);
    gemm(stream, ab, wihb, gb_ih, gi, NTOT, 768, OUTF, OUTF, 1, NTOT, NTOT, 0);
    gemm(stream, hb, whhb, gb_hh, gh, NTOT, 768, OUTF, OUTF, 1, NTOT, NTOT, 0);
    gru_update<<<cdiv(NF * OUTF, 256), 256, 0, stream>>>(gi, gh, h);
  }

  // ---- conv heads ----
  cvt_bf16<<<cdiv(NF * OUTF / 4, 256), 256, 0, stream>>>(h, hb, NF * OUTF);   // final h -> bf16
  concat_bf16<<<cdiv(NF * CATF, 256), 256, 0, stream>>>(h, feature, catb);

  // Y path: conv1 (k=3) -> pool 3/2 -> conv2 (k=1) -> pool 2/2
  gemm(stream, hb, c1p, c1b, y1, NB * 511, OUTF, OUTF, OUTF, 3, 511, NNODE, 1);
  maxpool_cl<<<cdiv(32L * 255 * (OUTF / 4), 256), 256, 0, stream>>>(y1, y1p, NB, 511, 255, OUTF, 3, 2);
  cvt_bf16<<<cdiv(32L * 255 * OUTF / 4, 256), 256, 0, stream>>>(y1p, y1pb, 32L * 255 * OUTF);
  gemm(stream, y1pb, c2p, c2b, y2, NB * 255, OUTF, OUTF, OUTF, 1, 255, 255, 1);
  maxpool_cl<<<cdiv(32L * 127 * (OUTF / 4), 256), 256, 0, stream>>>(y2, y2p, NB, 255, 127, OUTF, 2, 2);

  // Z path: convc1 (k=3) -> pool 3/2 -> convc2 (k=2) -> pool 2/2
  gemm(stream, catb, cc1p, cc1b, z1, NB * 511, CATF, CATF, CATF, 3, 511, NNODE, 1);
  maxpool_cl<<<cdiv(32L * 255 * (CATF / 4), 256), 256, 0, stream>>>(z1, z1p, NB, 511, 255, CATF, 3, 2);
  cvt_bf16<<<cdiv(32L * 255 * CATF / 4, 256), 256, 0, stream>>>(z1p, z1pb, 32L * 255 * CATF);
  gemm(stream, z1pb, cc2p, cc2b, z2, NB * 254, CATF, CATF, CATF, 2, 254, 255, 1);
  maxpool_cl<<<cdiv(32L * 127 * (CATF / 4), 256), 256, 0, stream>>>(z2, z2p, NB, 254, 127, CATF, 2, 2);

  // readout
  head_readout<<<NB, 128, 0, stream>>>(y2p, z2p, myw, myb, mzw, mzb, out);
}
